// Dilation_23493471109796
// MI455X (gfx1250) — compile-verified, matrix operand order checked by reading, not run on hardware
//
#include <hip/hip_runtime.h>
#include <hip/hip_bf16.h>

typedef __attribute__((ext_vector_type(16))) _Float16 v16h;
typedef __attribute__((ext_vector_type(8)))  float    v8f;

#define C_IN  64
#define KDIM  10
#define KK    (KDIM * KDIM)      // 100
#define PADW  5
#define HIN   256
#define HOUT  257
#define NBAT  8

// ---------------------------------------------------------------------------
// Kernel 1: S[o][p] = sum_c kernel[o][c][p]   (p = kh*10+kw, 0..99)
// Done as 16-row strips of row-sums via v_wmma_f32_16x16x32_f16 with an
// all-ones B matrix. 400 tasks = 4 strips x 100 tap positions, one wave each.
// ---------------------------------------------------------------------------
__global__ __launch_bounds__(32)
void Dilation_ksum_wmma(const float* __restrict__ kern, float* __restrict__ S)
{
    const int lane = threadIdx.x;        // 0..31 (wave32, EXEC all ones)
    const int row  = lane & 15;          // A-matrix M index
    const int half = lane >> 4;          // lane-group select
    const int task = blockIdx.x;         // 0..399
    const int o0   = (task & 3) * 16;    // output-channel strip base
    const int p    = task >> 2;          // tap position 0..99

    v8f acc = {};
    v16h bones;
#pragma unroll
    for (int e = 0; e < 16; ++e) bones[e] = (_Float16)1.0f;

#pragma unroll
    for (int chunk = 0; chunk < 2; ++chunk) {     // C=64 in two K=32 chunks
        v16h a;
#pragma unroll
        for (int e = 0; e < 16; ++e) {
            // 16-bit A 16x32 layout: VGPR v=e>>1, halfword w=e&1
            // lanes 0-15:  K = 2(v&3)+w (+16 if v>=4)
            // lanes 16-31: same + 8
            const int v = e >> 1, w = e & 1;
            const int k = 2 * (v & 3) + w + ((v >= 4) ? 16 : 0) + half * 8;
            const int c = chunk * 32 + k;
            a[e] = (_Float16)kern[(size_t)(o0 + row) * (C_IN * KK)
                                  + (size_t)c * KK + p];
        }
        acc = __builtin_amdgcn_wmma_f32_16x16x32_f16(
            /*neg_a=*/false, a, /*neg_b=*/false, bones,
            /*c_mod=*/(short)0, acc, /*reuse_a=*/false, /*reuse_b=*/false);
    }

    // D layout: lane 0 holds M=0..7 (N=0), lane 16 holds M=8..15 (N=0).
    if (row == 0) {
#pragma unroll
        for (int r = 0; r < 8; ++r)
            S[(size_t)(o0 + half * 8 + r) * KK + p] = acc[r];
    }
}

// ---------------------------------------------------------------------------
// Kernel 2: one block per (o, i) output row.
// T(o,i,j) = sum of S[o,kh,kw] over the window clipped to the image;
// out = (T > 0) ? 1 : 0, broadcast to all 8 batches (batch-invariant, and
// independent of x entirely since edge == exp(-5) is a positive constant).
// O(1) per pixel via an 11x11 2D inclusive prefix table in LDS.
// ---------------------------------------------------------------------------
__global__ __launch_bounds__(256)
void Dilation_write(const float* __restrict__ S, float* __restrict__ out)
{
    __shared__ float Sm[KK];
    __shared__ float P[KDIM + 1][KDIM + 1];

    const int o = blockIdx.x / HOUT;
    const int i = blockIdx.x % HOUT;

    if (threadIdx.x < KK)
        Sm[threadIdx.x] = S[(size_t)o * KK + threadIdx.x];
    __syncthreads();

    if (threadIdx.x == 0) {
        for (int a = 0; a <= KDIM; ++a) { P[0][a] = 0.0f; P[a][0] = 0.0f; }
        for (int kh = 0; kh < KDIM; ++kh)
            for (int kw = 0; kw < KDIM; ++kw)
                P[kh + 1][kw + 1] = Sm[kh * KDIM + kw]
                                  + P[kh][kw + 1] + P[kh + 1][kw] - P[kh][kw];
    }
    __syncthreads();

    // valid kh range for this output row: input r = i - PAD + kh in [0, 255]
    const int kh_lo = (PADW - i) > 0 ? (PADW - i) : 0;
    const int kh_hi_t = (HIN - 1) - (i - PADW);
    const int kh_hi = kh_hi_t < (KDIM - 1) ? kh_hi_t : (KDIM - 1);

    const size_t ostride = (size_t)HOUT * HOUT;          // 66049
    const size_t base    = (size_t)o * ostride + (size_t)i * HOUT;

    for (int j = threadIdx.x; j < HOUT; j += 256) {
        const int kw_lo = (PADW - j) > 0 ? (PADW - j) : 0;
        const int kw_hi_t = (HIN - 1) - (j - PADW);
        const int kw_hi = kw_hi_t < (KDIM - 1) ? kw_hi_t : (KDIM - 1);

        const float sum = P[kh_hi + 1][kw_hi + 1] - P[kh_lo][kw_hi + 1]
                        - P[kh_hi + 1][kw_lo]     + P[kh_lo][kw_lo];
        const float val = (sum > 0.0f) ? 1.0f : 0.0f;

        const size_t idx = base + (size_t)j;
#pragma unroll
        for (int n = 0; n < NBAT; ++n)
            __builtin_nontemporal_store(val, &out[idx + (size_t)n * C_IN * ostride]);
    }
}

// ---------------------------------------------------------------------------
// Launch: d_in[0] = x (unused — output provably independent of x),
//         d_in[1] = kernel (64*64*10*10 f32), d_out = 8*64*257*257 f32.
// d_ws holds S (64*100 f32 = 25.6 KB).
// ---------------------------------------------------------------------------
extern "C" void kernel_launch(void* const* d_in, const int* in_sizes, int n_in,
                              void* d_out, int out_size, void* d_ws, size_t ws_size,
                              hipStream_t stream)
{
    (void)in_sizes; (void)n_in; (void)out_size; (void)ws_size;
    const float* kern = (const float*)d_in[1];
    float* out = (float*)d_out;
    float* S   = (float*)d_ws;

    Dilation_ksum_wmma<<<4 * KK, 32, 0, stream>>>(kern, S);
    Dilation_write<<<C_IN * HOUT, 256, 0, stream>>>(S, out);
}
